// SNNv2_70866960384099
// MI455X (gfx1250) — matrix operand order repaired and checked by reading, not compiled, not run on hardware
//
#include <hip/hip_runtime.h>

// CDNA5 / gfx1250, wave32. bf16 hi/lo-split GEMMs on v_wmma_f32_16x16x32_bf16,
// with double-buffered async Global->LDS staging (ASYNCcnt) when available.

typedef __bf16 bf16_t;
typedef __attribute__((ext_vector_type(4)))  __bf16 v4bf;
typedef __attribute__((ext_vector_type(8)))  __bf16 v8bf;
typedef __attribute__((ext_vector_type(16))) __bf16 v16bf;
typedef __attribute__((ext_vector_type(8)))  float  v8f;
typedef __attribute__((ext_vector_type(4)))  int    v4i;

#define BETA 0.8f
#define THR  1.0f
#define LDSP 40   // padded LDS row stride (bf16 elems); multiple of 8 -> 16B-aligned vec ptrs

#if defined(__has_builtin)
# if __has_builtin(__builtin_amdgcn_global_load_async_to_lds_b128)
#  define ASYNC_LDS 1
# else
#  define ASYNC_LDS 0
# endif
#else
# define ASYNC_LDS 0
#endif

#if ASYNC_LDS
typedef __attribute__((address_space(1))) v4i* gptr_b128;   // global src (as int4*)
typedef __attribute__((address_space(3))) v4i* lptr_b128;   // LDS dst (as int4*)
#endif

__device__ __forceinline__ v8f wmma_bf16(v16bf a, v16bf b, v8f c) {
  // (neg_a, A, neg_b, B, c_mod, C, reuse_a, reuse_b)
  return __builtin_amdgcn_wmma_f32_16x16x32_bf16(false, a, false, b, (short)0, c,
                                                 false, false);
}

// 16-byte global -> LDS copy (async direct-to-LDS when supported).
__device__ __forceinline__ void cp16(const bf16_t* __restrict__ g, bf16_t* l) {
#if ASYNC_LDS
  __builtin_amdgcn_global_load_async_to_lds_b128((gptr_b128)g, (lptr_b128)l, 0, 0);
#else
  *(v8bf*)l = *(const v8bf*)g;
#endif
}

template <int N>
__device__ __forceinline__ void wait_async_chunk() {
#if ASYNC_LDS
# if __has_builtin(__builtin_amdgcn_s_wait_asynccnt)
  __builtin_amdgcn_s_wait_asynccnt(N);
# else
  asm volatile("s_wait_asynccnt %0" ::"i"(N));
# endif
#endif
}

// Stage ROWS x 32 bf16 chunk (row-major, leading dim ld) into padded LDS.
// Fully unrolled: (ROWS*4)/256 16B vectors per thread.
template <int ROWS>
__device__ __forceinline__ void stage(bf16_t* lds, const bf16_t* __restrict__ g,
                                      long ld, long row0, long k0, int tid) {
  constexpr int PER = (ROWS * 4) / 256;
#pragma unroll
  for (int it = 0; it < PER; ++it) {
    int v = tid + it * 256;
    int r = v >> 2, c = (v & 3) * 8;
    cp16(g + (row0 + r) * ld + k0 + c, lds + r * LDSP + c);
  }
}

// A fragment 16x32 bf16: lane<16 -> M=lane, K={0..7,16..23}; lane>=16 -> K={8..15,24..31}
__device__ __forceinline__ v16bf frag_a(const bf16_t* lds, int mrow, int lane) {
  const int half = (lane < 16) ? 0 : 8;
  const bf16_t* p = lds + (mrow + (lane & 15)) * LDSP + half;
  v8bf x0 = *(const v8bf*)p;
  v8bf x1 = *(const v8bf*)(p + 16);
  v16bf r;
#pragma unroll
  for (int i = 0; i < 8; ++i) { r[i] = x0[i]; r[8 + i] = x1[i]; }
  return r;
}

// B fragment 32x16 bf16 (B^T rows = W rows): lane<16 -> N=lane, K={0..7,16..23}; lane>=16 -> K={8..15,24..31}
__device__ __forceinline__ v16bf frag_b(const bf16_t* lds, int nrow, int lane) {
  const int koff = (lane < 16) ? 0 : 8;
  const bf16_t* p = lds + (nrow + (lane & 15)) * LDSP + koff;
  v8bf x0 = *(const v8bf*)p;
  v8bf x1 = *(const v8bf*)(p + 16);
  v16bf r;
#pragma unroll
  for (int i = 0; i < 8; ++i) { r[i] = x0[i]; r[8 + i] = x1[i]; }
  return r;
}

// ---------------- fp32 -> (hi, lo) bf16 split, 4 elems / thread ----------------
__global__ __launch_bounds__(256) void split_kernel(const float* __restrict__ in,
                                                    bf16_t* __restrict__ hi,
                                                    bf16_t* __restrict__ lo, long n4) {
  long i = (long)blockIdx.x * blockDim.x + threadIdx.x;
  if (i >= n4) return;
  float4 v = ((const float4*)in)[i];
  float a[4] = {v.x, v.y, v.z, v.w};
  v4bf h, l;
#pragma unroll
  for (int k = 0; k < 4; ++k) {
    bf16_t hh = (bf16_t)a[k];            // RNE fptrunc
    h[k] = hh;
    l[k] = (bf16_t)(a[k] - (float)hh);   // residual
  }
  ((v4bf*)hi)[i] = h;
  ((v4bf*)lo)[i] = l;
}

// ---------------- LIF layer-1 step: update m1, emit s1 (exact bf16) ----------------
__global__ __launch_bounds__(256) void lif1_kernel(const float* __restrict__ cur1,
                                                   float* m1,
                                                   bf16_t* __restrict__ s1, long n4) {
  long i = (long)blockIdx.x * blockDim.x + threadIdx.x;
  if (i >= n4) return;
  float4 mv = ((const float4*)m1)[i];
  float4 cv = ((const float4*)cur1)[i];
  float m[4] = {mv.x, mv.y, mv.z, mv.w};
  float c[4] = {cv.x, cv.y, cv.z, cv.w};
  v4bf s;
#pragma unroll
  for (int k = 0; k < 4; ++k) {
    float r = (m[k] > THR) ? THR : 0.0f;   // reset from previous membrane
    m[k] = BETA * m[k] + c[k] - r;
    s[k] = (bf16_t)((m[k] > THR) ? 1.0f : 0.0f);
  }
  float4 mo; mo.x = m[0]; mo.y = m[1]; mo.z = m[2]; mo.w = m[3];
  ((float4*)m1)[i] = mo;
  ((v4bf*)s1)[i] = s;
}

// ---------------- fc1: cur1 = x @ W1^T + b1  (3-product hi/lo split) ----------------
__global__ __launch_bounds__(256) void gemm_fc1(
    const bf16_t* __restrict__ xhi, const bf16_t* __restrict__ xlo,
    const bf16_t* __restrict__ w1hi, const bf16_t* __restrict__ w1lo,
    const float* __restrict__ b1, float* __restrict__ cur1) {
  const long K = 2048, N = 4096;
  constexpr int ABLK = 128 * LDSP, BBLK = 64 * LDSP;
  constexpr int BUF = 2 * ABLK + 2 * BBLK;      // Ah, Al, Bh, Bl
  __shared__ __align__(16) bf16_t sm[2 * BUF];  // ping-pong, 61.4 KB
  const int tid = threadIdx.x, lane = tid & 31, wid = tid >> 5;
  const int wm = (wid & 3) * 32, wn = (wid >> 2) * 32;
  const long m0 = (long)blockIdx.y * 128, n0 = (long)blockIdx.x * 64;

  auto issue = [&](int pb, long k0) {
    bf16_t* b = sm + pb * BUF;
    stage<128>(b,                   xhi,  K, m0, k0, tid);
    stage<128>(b + ABLK,            xlo,  K, m0, k0, tid);
    stage<64>(b + 2 * ABLK,         w1hi, K, n0, k0, tid);
    stage<64>(b + 2 * ABLK + BBLK,  w1lo, K, n0, k0, tid);
  };

  v8f acc[2][2] = {};
  issue(0, 0);
  const int NK = (int)(K / 32);
  for (int ki = 0; ki < NK; ++ki) {
    const int cur = ki & 1;
    if (ki + 1 < NK) {                 // prefetch next chunk into the other buffer
      issue(1 - cur, (long)(ki + 1) * 32);
      wait_async_chunk<6>();           // 6 ops just issued -> previous chunk complete
    } else {
      wait_async_chunk<0>();
    }
    __syncthreads();
    bf16_t* b = sm + cur * BUF;
    v16bf ah[2], al[2], bh[2], bl[2];
#pragma unroll
    for (int i = 0; i < 2; ++i) {
      ah[i] = frag_a(b, wm + i * 16, lane);
      al[i] = frag_a(b + ABLK, wm + i * 16, lane);
      bh[i] = frag_b(b + 2 * ABLK, wn + i * 16, lane);
      bl[i] = frag_b(b + 2 * ABLK + BBLK, wn + i * 16, lane);
    }
#pragma unroll
    for (int i = 0; i < 2; ++i)
#pragma unroll
      for (int j = 0; j < 2; ++j) {
        acc[i][j] = wmma_bf16(ah[i], bh[j], acc[i][j]);   // hi*hi
        acc[i][j] = wmma_bf16(ah[i], bl[j], acc[i][j]);   // hi*lo
        acc[i][j] = wmma_bf16(al[i], bh[j], acc[i][j]);   // lo*hi
      }
    __syncthreads();
  }
  const int ncol = lane & 15, mhalf = (lane < 16) ? 0 : 8;
#pragma unroll
  for (int i = 0; i < 2; ++i)
#pragma unroll
    for (int j = 0; j < 2; ++j) {
      long gn = n0 + wn + j * 16 + ncol;
      float bias = b1[gn];
#pragma unroll
      for (int v = 0; v < 8; ++v) {
        long gm = m0 + wm + i * 16 + mhalf + v;
        cur1[gm * N + gn] = acc[i][j][v] + bias;
      }
    }
}

// -------- fc2 + LIF2 fused: cur2 = s1 @ W2^T + b2, then membrane update + record --------
__global__ __launch_bounds__(256) void gemm_fc2_lif(
    const bf16_t* __restrict__ s1, const bf16_t* __restrict__ w2hi,
    const bf16_t* __restrict__ w2lo, const float* __restrict__ b2,
    float* m2, float* __restrict__ spk_out, float* __restrict__ mem_out) {
  const long K = 4096, N = 1024;
  constexpr int ABLK = 128 * LDSP, BBLK = 64 * LDSP;
  constexpr int BUF = ABLK + 2 * BBLK;          // A, Bh, Bl
  __shared__ __align__(16) bf16_t sm[2 * BUF];  // ping-pong, 40 KB
  const int tid = threadIdx.x, lane = tid & 31, wid = tid >> 5;
  const int wm = (wid & 3) * 32, wn = (wid >> 2) * 32;
  const long m0 = (long)blockIdx.y * 128, n0 = (long)blockIdx.x * 64;

  auto issue = [&](int pb, long k0) {
    bf16_t* b = sm + pb * BUF;
    stage<128>(b,              s1,   K, m0, k0, tid);
    stage<64>(b + ABLK,        w2hi, K, n0, k0, tid);
    stage<64>(b + ABLK + BBLK, w2lo, K, n0, k0, tid);
  };

  v8f acc[2][2] = {};
  issue(0, 0);
  const int NK = (int)(K / 32);
  for (int ki = 0; ki < NK; ++ki) {
    const int cur = ki & 1;
    if (ki + 1 < NK) {
      issue(1 - cur, (long)(ki + 1) * 32);
      wait_async_chunk<4>();           // 4 ops just issued -> previous chunk complete
    } else {
      wait_async_chunk<0>();
    }
    __syncthreads();
    bf16_t* b = sm + cur * BUF;
    v16bf a[2], bh[2], bl[2];
#pragma unroll
    for (int i = 0; i < 2; ++i) {
      a[i]  = frag_a(b, wm + i * 16, lane);
      bh[i] = frag_b(b + ABLK, wn + i * 16, lane);
      bl[i] = frag_b(b + ABLK + BBLK, wn + i * 16, lane);
    }
#pragma unroll
    for (int i = 0; i < 2; ++i)
#pragma unroll
      for (int j = 0; j < 2; ++j) {
        acc[i][j] = wmma_bf16(a[i], bh[j], acc[i][j]);    // s1 * W2_hi
        acc[i][j] = wmma_bf16(a[i], bl[j], acc[i][j]);    // s1 * W2_lo
      }
    __syncthreads();
  }
  const int ncol = lane & 15, mhalf = (lane < 16) ? 0 : 8;
#pragma unroll
  for (int i = 0; i < 2; ++i)
#pragma unroll
    for (int j = 0; j < 2; ++j) {
      long gn = n0 + wn + j * 16 + ncol;
      float bias = b2[gn];
#pragma unroll
      for (int v = 0; v < 8; ++v) {
        long gm = m0 + wm + i * 16 + mhalf + v;
        long idx = gm * N + gn;
        float c = acc[i][j][v] + bias;
        float m = m2[idx];
        float r = (m > THR) ? THR : 0.0f;
        m = BETA * m + c - r;
        spk_out[idx] = (m > THR) ? 1.0f : 0.0f;
        mem_out[idx] = m;
        m2[idx] = m;
      }
    }
}

// ---------------- host launch ----------------
extern "C" void kernel_launch(void* const* d_in, const int* in_sizes, int n_in,
                              void* d_out, int out_size, void* d_ws, size_t ws_size,
                              hipStream_t stream) {
  (void)in_sizes; (void)n_in; (void)out_size; (void)ws_size;
  const float* x  = (const float*)d_in[0];
  const float* W1 = (const float*)d_in[1];
  const float* b1 = (const float*)d_in[2];
  const float* W2 = (const float*)d_in[3];
  const float* b2 = (const float*)d_in[4];

  const long B = 4096, DIN = 2048, H = 4096, O = 1024, T = 20;

  // Workspace carve-out (256 MB total, all regions 16B-aligned).
  char* p = (char*)d_ws;
  bf16_t* xhi  = (bf16_t*)p; p += B * DIN * 2;
  bf16_t* xlo  = (bf16_t*)p; p += B * DIN * 2;
  bf16_t* w1hi = (bf16_t*)p; p += H * DIN * 2;
  bf16_t* w1lo = (bf16_t*)p; p += H * DIN * 2;
  bf16_t* w2hi = (bf16_t*)p; p += O * H * 2;
  bf16_t* w2lo = (bf16_t*)p; p += O * H * 2;
  float*  cur1 = (float*)p;  p += B * H * 4;
  float*  m1   = (float*)p;  p += B * H * 4;
  bf16_t* s1   = (bf16_t*)p; p += B * H * 2;
  float*  m2   = (float*)p;  p += B * O * 4;

  (void)hipMemsetAsync(m1, 0, B * H * 4, stream);
  (void)hipMemsetAsync(m2, 0, B * O * 4, stream);

  split_kernel<<<(unsigned)(B * DIN / 4 / 256), 256, 0, stream>>>(x, xhi, xlo, B * DIN / 4);
  split_kernel<<<(unsigned)(H * DIN / 4 / 256), 256, 0, stream>>>(W1, w1hi, w1lo, H * DIN / 4);
  split_kernel<<<(unsigned)(O * H / 4 / 256),   256, 0, stream>>>(W2, w2hi, w2lo, O * H / 4);

  gemm_fc1<<<dim3((unsigned)(H / 64), (unsigned)(B / 128)), 256, 0, stream>>>(
      xhi, xlo, w1hi, w1lo, b1, cur1);

  float* spk = (float*)d_out;
  float* mem = spk + T * B * O;
  for (int t = 0; t < T; ++t) {
    lif1_kernel<<<(unsigned)(B * H / 4 / 256), 256, 0, stream>>>(cur1, m1, s1, B * H / 4);
    gemm_fc2_lif<<<dim3((unsigned)(O / 64), (unsigned)(B / 128)), 256, 0, stream>>>(
        s1, w2hi, w2lo, b2, m2, spk + (long)t * B * O, mem + (long)t * B * O);
  }
}
